// TransformerEncoder_75883482186451
// MI455X (gfx1250) — matrix operand order repaired and checked by reading, not compiled, not run on hardware
//
#include <hip/hip_runtime.h>
#include <math.h>

// ---------------------------------------------------------------- constants
#define DMODEL   1024
#define NHEADS   16
#define HEAD_DIM 64
#define FFN_DIM  4096
#define BATCH    2
#define SEQ      2048
#define NROWS    (BATCH * SEQ)          // 4096
static constexpr float ATTN_SCALE = 0.125f;   // HEAD_DIM^-0.5

// ---------------------------------------------------------------- types
typedef __attribute__((ext_vector_type(16))) __bf16 v16bf;
typedef __attribute__((ext_vector_type(8)))  __bf16 v8bf;
typedef __attribute__((ext_vector_type(8)))  float  v8f;
typedef int v4i __attribute__((vector_size(4 * sizeof(int))));

// ------------------------------------------------ gfx1250 async global->LDS
#if defined(__HIP_DEVICE_COMPILE__) && \
    __has_builtin(__builtin_amdgcn_global_load_async_to_lds_b128)
#define USE_ASYNC_LDS 1
#else
#define USE_ASYNC_LDS 0
#endif

#if USE_ASYNC_LDS
typedef __attribute__((address_space(1))) v4i* g_v4i_p;
typedef __attribute__((address_space(3))) v4i* l_v4i_p;
// 16-byte async copy global->LDS, tracked by ASYNCcnt
__device__ inline void async_cp16(const void* g, void* l) {
  __builtin_amdgcn_global_load_async_to_lds_b128((g_v4i_p)g, (l_v4i_p)l, 0, 0);
}
__device__ inline void wait_async0() {
#if __has_builtin(__builtin_amdgcn_s_wait_asynccnt)
  __builtin_amdgcn_s_wait_asynccnt(0);
#else
  asm volatile("s_wait_asynccnt 0x0" ::: "memory");
#endif
}
#endif

// D = A(16x32 bf16) * B(32x16 bf16) + C(16x16 f32)
__device__ inline v8f wmma_bf16(v16bf a, v16bf b, v8f c) {
  return __builtin_amdgcn_wmma_f32_16x16x32_bf16(
      /*neg_a=*/false, a, /*neg_b=*/false, b,
      /*c_mod=*/(short)0, c, /*reuse_a=*/false, /*reuse_b=*/false);
}

// A fragment (16x32, MxK), rows striped per lane (ISA 7.12.2):
// lane<16 : elems 0..7 = K[k0+0..7],  elems 8..15 = K[k0+16..23]
// lane>=16: elems 0..7 = K[k0+8..15], elems 8..15 = K[k0+24..31]
__device__ inline v16bf load_a_frag(const __bf16* base, int ld, int row0,
                                    int k0, int lane) {
  const int r  = row0 + (lane & 15);
  const int kh = (lane >> 4) * 8;
  const __bf16* p = base + (size_t)r * ld + k0;
  union { v16bf v; v8bf h[2]; } u;
  u.h[0] = *(const v8bf*)(p + kh);
  u.h[1] = *(const v8bf*)(p + 16 + kh);
  return u.v;
}

// B fragment (32x16, KxN) from an LDS tile stored [N][K] (column-contiguous):
// lane n holds column (col0 + n%16); lane<16 K[k0+0..7,16..23], lane>=16 K[k0+8..15,24..31]
__device__ inline v16bf load_b_frag(const __bf16* base, int ld, int col0,
                                    int k0, int lane) {
  const int c  = col0 + (lane & 15);
  const int kq = (lane >> 4) * 8;
  const __bf16* p = base + (size_t)c * ld + k0 + kq;
  union { v16bf v; v8bf h[2]; } u;
  u.h[0] = *(const v8bf*)(p);
  u.h[1] = *(const v8bf*)(p + 16);
  return u.v;
}

// ---------------------------------------------------------------- f32 -> bf16
__global__ __launch_bounds__(256)
void enc_f2bf_kernel(const float* __restrict__ in, __bf16* __restrict__ out, int n) {
  int i = (blockIdx.x * 256 + threadIdx.x) * 4;
  if (i + 3 < n) {
    float4 f = *(const float4*)(in + i);
    out[i + 0] = (__bf16)f.x;
    out[i + 1] = (__bf16)f.y;
    out[i + 2] = (__bf16)f.z;
    out[i + 3] = (__bf16)f.w;
  }
}

// ---------------------------------------------------------------- layernorm
__device__ inline float block_reduce_sum(float s, float* red, int t) {
  #pragma unroll
  for (int m = 16; m > 0; m >>= 1) s += __shfl_xor(s, m, 32);
  const int wid = t >> 5;
  if ((t & 31) == 0) red[wid] = s;
  __syncthreads();
  if (t < 8) {
    float x = red[t];
    #pragma unroll
    for (int m = 4; m > 0; m >>= 1) x += __shfl_xor(x, m, 32);
    if (t == 0) red[8] = x;
  }
  __syncthreads();
  float r = red[8];
  __syncthreads();
  return r;
}

// y = LN(a (+ res)) * gamma + beta ; optional bf16 copy of y
template <bool WRITE_BF16, bool HAS_RES>
__global__ __launch_bounds__(256)
void enc_ln_kernel(const float* __restrict__ a, const float* __restrict__ res,
                   const float* __restrict__ gamma, const float* __restrict__ beta,
                   float* __restrict__ outF, __bf16* __restrict__ outB) {
  __shared__ float red[16];
  const int row = blockIdx.x;
  const int t = threadIdx.x;
  const float* ap = a + (size_t)row * DMODEL;
  const float* rp = HAS_RES ? (res + (size_t)row * DMODEL) : nullptr;

  float v[4];
  float s = 0.f;
  #pragma unroll
  for (int i = 0; i < 4; ++i) {
    const int idx = t + i * 256;
    v[i] = ap[idx] + (HAS_RES ? rp[idx] : 0.f);
    s += v[i];
  }
  const float mean = block_reduce_sum(s, red, t) * (1.0f / DMODEL);
  float s2 = 0.f;
  #pragma unroll
  for (int i = 0; i < 4; ++i) { float d = v[i] - mean; s2 += d * d; }
  const float var = block_reduce_sum(s2, red, t) * (1.0f / DMODEL);
  const float rstd = rsqrtf(var + 1e-5f);
  #pragma unroll
  for (int i = 0; i < 4; ++i) {
    const int idx = t + i * 256;
    const float y = (v[i] - mean) * rstd * gamma[idx] + beta[idx];
    outF[(size_t)row * DMODEL + idx] = y;
    if (WRITE_BF16) outB[(size_t)row * DMODEL + idx] = (__bf16)y;
  }
}

// ---------------------------------------------------------------- WMMA GEMM
// C[M,N] = A[M,K] * W[N,K]^T + bias.  EPI: 0 = f32 out, 1 = bf16 out, 2 = gelu->bf16
// Block tile 64(M) x 256(N), BK=32; 8 waves in 2x4 grid, each 32x64 (2x4 WMMA
// tiles -> 8 wmma per 6 fragment loads per k-step).  LDS double-buffered,
// global->LDS staged via async copies when available.
template <int EPI>
__global__ __launch_bounds__(256)
void enc_gemm_kernel(const __bf16* __restrict__ A, const __bf16* __restrict__ W,
                     const float* __restrict__ bias,
                     float* __restrict__ outF, __bf16* __restrict__ outB,
                     int M, int N, int K) {
  __shared__ __bf16 As[2][64][40];     // [M][K] + pad   (10 KB)
  __shared__ __bf16 Bs[2][256][40];    // [N][K] + pad   (40 KB)

  const int t = threadIdx.x;
  const int lane = t & 31, wid = t >> 5;
  const int bm = blockIdx.y, bn = blockIdx.x;
  const int m0 = (wid >> 2) * 32;   // 0 or 32
  const int n0 = (wid & 3) * 64;    // 0..192

  v8f c[2][4] = {};
  const int ar = t >> 2, ac = (t & 3) * 8;   // 64 rows x (4 x 8 elems)
  const __bf16* Ag = A + (size_t)(bm * 64)  * K + (size_t)ar * K + ac;
  const __bf16* Wg = W + (size_t)(bn * 256) * K + (size_t)ar * K + ac;

#if USE_ASYNC_LDS
  auto gcopy = [&](int buf, int kt) {
    async_cp16(Ag + kt * 32, &As[buf][ar][ac]);
    #pragma unroll
    for (int q = 0; q < 4; ++q)
      async_cp16(Wg + (size_t)(q * 64) * K + kt * 32, &Bs[buf][ar + q * 64][ac]);
  };
  gcopy(0, 0);
  wait_async0();
  __syncthreads();
#else
  v8bf aReg;
  v8bf bReg[4];
  auto gload = [&](int kt) {
    aReg = *(const v8bf*)(Ag + kt * 32);
    #pragma unroll
    for (int q = 0; q < 4; ++q)
      bReg[q] = *(const v8bf*)(Wg + (size_t)(q * 64) * K + kt * 32);
  };
  auto sstore = [&](int buf) {
    *(v8bf*)&As[buf][ar][ac] = aReg;
    #pragma unroll
    for (int q = 0; q < 4; ++q)
      *(v8bf*)&Bs[buf][ar + q * 64][ac] = bReg[q];
  };
  gload(0);
  sstore(0);
  __syncthreads();
#endif

  const int nk = K >> 5;
  for (int kt = 0; kt < nk; ++kt) {
    const int cur = kt & 1;
    const bool pf = (kt + 1 < nk);
#if USE_ASYNC_LDS
    if (pf) gcopy(cur ^ 1, kt + 1);
#else
    if (pf) gload(kt + 1);
#endif

    v16bf a[2], b[4];
    #pragma unroll
    for (int i = 0; i < 2; ++i)
      a[i] = load_a_frag(&As[cur][0][0], 40, m0 + i * 16, 0, lane);
    #pragma unroll
    for (int j = 0; j < 4; ++j)
      b[j] = load_b_frag(&Bs[cur][0][0], 40, n0 + j * 16, 0, lane);
    #pragma unroll
    for (int i = 0; i < 2; ++i)
      #pragma unroll
      for (int j = 0; j < 4; ++j)
        c[i][j] = wmma_bf16(a[i], b[j], c[i][j]);

#if USE_ASYNC_LDS
    if (pf) wait_async0();       // own copies LDS-resident before the barrier
#else
    if (pf) sstore(cur ^ 1);
#endif
    __syncthreads();
  }

  // Epilogue.  C layout: lane holds col (n%16), rows e + 8*(lane/16).
  const int colBase = bn * 256 + n0 + (lane & 15);
  const int rowSub  = (lane >> 4) * 8;
  #pragma unroll
  for (int i = 0; i < 2; ++i) {
    #pragma unroll
    for (int j = 0; j < 4; ++j) {
      const int col = colBase + j * 16;
      const float bv = bias[col];
      #pragma unroll
      for (int e = 0; e < 8; ++e) {
        const int row = bm * 64 + m0 + i * 16 + rowSub + e;
        float v = c[i][j][e] + bv;
        if (EPI == 1) {
          outB[(size_t)row * N + col] = (__bf16)v;
        } else if (EPI == 2) {
          float g = 0.5f * v * (1.0f + erff(v * 0.70710678118f));
          outB[(size_t)row * N + col] = (__bf16)g;
        } else {
          outF[(size_t)row * N + col] = v;
        }
      }
    }
  }
}

// ---------------------------------------------------------------- flash attention
// qkv: [NROWS, 3*DMODEL] bf16 (Q | K | V), out: [NROWS, DMODEL] f32
// Block: 128 threads (4 waves), 64-query tile of one (b,h); streams 64-key tiles.
__global__ __launch_bounds__(128)
void enc_attn_kernel(const __bf16* __restrict__ qkv, float* __restrict__ out) {
  __shared__ __bf16 Qs[64][72];        // [q][d]
  __shared__ __bf16 Ks[64][72];        // [k][d]
  __shared__ __bf16 Vt[64][72];        // [d][k]  (transposed for PV B-frags)
  __shared__ __bf16 Ps[4][16][72];     // per-wave P staging [q][k]

  const int t = threadIdx.x, lane = t & 31, wid = t >> 5;
  const int b = blockIdx.y >> 4;
  const int h = blockIdx.y & 15;
  const int q0 = blockIdx.x * 64;
  const size_t rowBase = (size_t)b * SEQ;
  const int ldq = 3 * DMODEL;

  // Load Q tile once (synced by the barrier inside the first loop iteration).
  {
    const __bf16* Qg = qkv + (rowBase + q0) * (size_t)ldq + h * HEAD_DIM;
    #pragma unroll
    for (int c4 = 0; c4 < 4; ++c4) {
      const int r  = (t >> 3) + c4 * 16;
      const int d0 = (t & 7) * 8;
      *(v8bf*)&Qs[r][d0] = *(const v8bf*)(Qg + (size_t)r * ldq + d0);
    }
  }

  v8f o[4] = {};
  float mrow[8], lrow[8];
  #pragma unroll
  for (int e = 0; e < 8; ++e) { mrow[e] = -1e30f; lrow[e] = 0.f; }

  const __bf16* Kg = qkv + rowBase * (size_t)ldq + DMODEL     + h * HEAD_DIM;
  const __bf16* Vg = qkv + rowBase * (size_t)ldq + 2 * DMODEL + h * HEAD_DIM;

  for (int kt = 0; kt < SEQ / 64; ++kt) {
    __syncthreads();   // previous tile's WMMA reads of Ks/Vt complete
    #pragma unroll
    for (int c4 = 0; c4 < 4; ++c4) {
      const int r  = (t >> 3) + c4 * 16;
      const int d0 = (t & 7) * 8;
      const size_t goff = (size_t)(kt * 64 + r) * ldq + d0;
#if USE_ASYNC_LDS
      async_cp16(Kg + goff, &Ks[r][d0]);
#else
      *(v8bf*)&Ks[r][d0] = *(const v8bf*)(Kg + goff);
#endif
      v8bf vv = *(const v8bf*)(Vg + goff);
      #pragma unroll
      for (int i = 0; i < 8; ++i) Vt[d0 + i][r] = vv[i];
    }
#if USE_ASYNC_LDS
    wait_async0();
#endif
    __syncthreads();

    // S = Q * K^T  (reduce over head_dim: two K=32 WMMA steps)
    v8f s[4] = {};
    #pragma unroll
    for (int d0 = 0; d0 < 64; d0 += 32) {
      v16bf aq = load_a_frag(&Qs[0][0], 72, wid * 16, d0, lane);
      #pragma unroll
      for (int j = 0; j < 4; ++j) {
        v16bf bk = load_b_frag(&Ks[0][0], 72, j * 16, d0, lane);
        s[j] = wmma_bf16(aq, bk, s[j]);
      }
    }

    // Online softmax. Row e+8*(lane/16) is spread over the 16 lanes of a half.
    float p[4][8];
    #pragma unroll
    for (int e = 0; e < 8; ++e) {
      float mx = fmaxf(fmaxf(s[0][e], s[1][e]), fmaxf(s[2][e], s[3][e])) * ATTN_SCALE;
      #pragma unroll
      for (int msk = 1; msk < 16; msk <<= 1) mx = fmaxf(mx, __shfl_xor(mx, msk, 32));
      const float mnew  = fmaxf(mrow[e], mx);
      const float alpha = __expf(mrow[e] - mnew);
      float ssum = 0.f;
      #pragma unroll
      for (int j = 0; j < 4; ++j) {
        const float pe = __expf(s[j][e] * ATTN_SCALE - mnew);
        p[j][e] = pe;
        ssum += pe;
      }
      #pragma unroll
      for (int msk = 1; msk < 16; msk <<= 1) ssum += __shfl_xor(ssum, msk, 32);
      lrow[e] = lrow[e] * alpha + ssum;
      mrow[e] = mnew;
      #pragma unroll
      for (int j = 0; j < 4; ++j) o[j][e] *= alpha;
    }

    // Re-lay P (C layout) into [q][k] in this wave's own LDS region.
    const int pr = (lane >> 4) * 8;
    const int pc = lane & 15;
    #pragma unroll
    for (int j = 0; j < 4; ++j)
      #pragma unroll
      for (int e = 0; e < 8; ++e)
        Ps[wid][pr + e][j * 16 + pc] = (__bf16)p[j][e];

    // O += P * V  (reduce over 64 keys: two K=32 WMMA steps)
    #pragma unroll
    for (int kc = 0; kc < 64; kc += 32) {
      v16bf ap = load_a_frag(&Ps[wid][0][0], 72, 0, kc, lane);
      #pragma unroll
      for (int j = 0; j < 4; ++j) {
        v16bf bv = load_b_frag(&Vt[0][0], 72, j * 16, kc, lane);
        o[j] = wmma_bf16(ap, bv, o[j]);
      }
    }
  }

  // Normalize and write [b, s, h*64 + d].
  const int pr = (lane >> 4) * 8;
  const int pc = lane & 15;
  #pragma unroll
  for (int j = 0; j < 4; ++j)
    #pragma unroll
    for (int e = 0; e < 8; ++e) {
      const size_t row = rowBase + q0 + wid * 16 + pr + e;
      const int col = h * HEAD_DIM + j * 16 + pc;
      out[row * DMODEL + col] = o[j][e] / lrow[e];
    }
}

// ---------------------------------------------------------------- launch
extern "C" void kernel_launch(void* const* d_in, const int* in_sizes, int n_in,
                              void* d_out, int out_size, void* d_ws, size_t ws_size,
                              hipStream_t stream) {
  (void)in_sizes; (void)n_in; (void)out_size; (void)ws_size;
  const float* src   = (const float*)d_in[0];
  const float* pre_g = (const float*)d_in[1];
  const float* pre_b = (const float*)d_in[2];
  const float* qkv_w = (const float*)d_in[3];
  const float* qkv_b = (const float*)d_in[4];
  const float* an_g  = (const float*)d_in[5];
  const float* an_b  = (const float*)d_in[6];
  const float* fc1_w = (const float*)d_in[7];
  const float* fc1_b = (const float*)d_in[8];
  const float* fc2_w = (const float*)d_in[9];
  const float* fc2_b = (const float*)d_in[10];

  char* ws = (char*)d_ws;
  size_t off = 0;
  auto alloc = [&](size_t bytes) {
    off = (off + 255) & ~(size_t)255;
    void* p = ws + off;
    off += bytes;
    return p;
  };

  float*  xF   = (float*) alloc((size_t)NROWS * DMODEL * 4);
  __bf16* xB   = (__bf16*)alloc((size_t)NROWS * DMODEL * 2);
  __bf16* qkvB = (__bf16*)alloc((size_t)NROWS * 3 * DMODEL * 2);
  float*  attn = (float*) alloc((size_t)NROWS * DMODEL * 4);
  float*  x2F  = (float*) alloc((size_t)NROWS * DMODEL * 4);
  __bf16* x2B  = (__bf16*)alloc((size_t)NROWS * DMODEL * 2);
  __bf16* hB   = (__bf16*)alloc((size_t)NROWS * FFN_DIM * 2);
  float*  ffn  = (float*) alloc((size_t)NROWS * DMODEL * 4);
  __bf16* wqkv = (__bf16*)alloc((size_t)3 * DMODEL * DMODEL * 2);
  __bf16* wfc1 = (__bf16*)alloc((size_t)FFN_DIM * DMODEL * 2);
  __bf16* wfc2 = (__bf16*)alloc((size_t)DMODEL * FFN_DIM * 2);

  // Weight conversions (fp32 -> bf16)
  enc_f2bf_kernel<<<(3 * DMODEL * DMODEL) / 1024, 256, 0, stream>>>(qkv_w, wqkv, 3 * DMODEL * DMODEL);
  enc_f2bf_kernel<<<(FFN_DIM * DMODEL) / 1024, 256, 0, stream>>>(fc1_w, wfc1, FFN_DIM * DMODEL);
  enc_f2bf_kernel<<<(DMODEL * FFN_DIM) / 1024, 256, 0, stream>>>(fc2_w, wfc2, DMODEL * FFN_DIM);

  // 1) PreNorm: x = LN(src)
  enc_ln_kernel<true, false><<<NROWS, 256, 0, stream>>>(src, nullptr, pre_g, pre_b, xF, xB);
  // 2) QKV = x @ qkv_w^T + qkv_b  (bf16 out)
  enc_gemm_kernel<1><<<dim3(3 * DMODEL / 256, NROWS / 64), 256, 0, stream>>>(
      xB, wqkv, qkv_b, nullptr, qkvB, NROWS, 3 * DMODEL, DMODEL);
  // 3) attention
  enc_attn_kernel<<<dim3(SEQ / 64, BATCH * NHEADS), 128, 0, stream>>>(qkvB, attn);
  // 4) AddNorm #1: x2 = LN(attn + x)
  enc_ln_kernel<true, true><<<NROWS, 256, 0, stream>>>(attn, xF, an_g, an_b, x2F, x2B);
  // 5) h = gelu(x2 @ fc1_w^T + fc1_b)  (bf16 out)
  enc_gemm_kernel<2><<<dim3(FFN_DIM / 256, NROWS / 64), 256, 0, stream>>>(
      x2B, wfc1, fc1_b, nullptr, hB, NROWS, FFN_DIM, DMODEL);
  // 6) ffn = h @ fc2_w^T + fc2_b  (f32 out)
  enc_gemm_kernel<0><<<dim3(DMODEL / 256, NROWS / 64), 256, 0, stream>>>(
      hB, wfc2, fc2_b, ffn, nullptr, NROWS, DMODEL, FFN_DIM);
  // 7) AddNorm #2: out = LN(ffn + x2)
  enc_ln_kernel<false, true><<<NROWS, 256, 0, stream>>>(ffn, x2F, an_g, an_b, (float*)d_out, nullptr);
}
